// VectorQuantizer_16114717294699
// MI455X (gfx1250) — matrix operand order repaired and checked by reading, not compiled, not run on hardware
//
#include <hip/hip_runtime.h>

typedef __attribute__((ext_vector_type(16))) __bf16 v16bf;
typedef __attribute__((ext_vector_type(8)))  float  v8f;
typedef __attribute__((ext_vector_type(4)))  unsigned int v4u;
typedef __attribute__((ext_vector_type(8)))  int    v8i;
typedef __attribute__((ext_vector_type(4)))  int    v4i;
typedef unsigned short u16;
typedef unsigned int   u32;
typedef unsigned long long u64;

#define B_  16
#define D_  256
#define T_  1024
#define K_  8192
#define N_  (B_ * T_)          // 16384
#define TILE_C 32              // codebook rows per LDS buffer (double-buffered)
#define NSTAGE (K_ / TILE_C)   // 256 pipeline stages
#define LDS_STRIDE 264         // u16 per code row in LDS (256 + 8 pad -> no bank conflicts)
#define WAVES 8                // waves per workgroup in main kernel

// ---------------- ws layout (bytes) ----------------
#define XB_OFF     0u                           // x_bf16: N*D*2     = 8388608
#define EB_OFF     8388608u                     // emb_bf16: K*D*2   = 4194304
#define ENORM_OFF  12582912u                    // |e|^2: K*4        = 32768
#define IDX_OFF    12615680u                    // idx: N*4          = 65536
#define CNT_OFF    12681216u                    // counts: K*4       = 32768
#define LOSS_OFF   12713984u                    // loss accum: 4

#if defined(__has_builtin)
#if __has_builtin(__builtin_amdgcn_tensor_load_to_lds) && \
    __has_builtin(__builtin_amdgcn_s_wait_tensorcnt)
#define USE_TDM 1
#endif
#endif

__device__ __forceinline__ u16 f2bf(float f) {
    u32 u = __float_as_uint(f);
    u32 r = u + 0x7FFFu + ((u >> 16) & 1u);     // round-to-nearest-even
    return (u16)(r >> 16);
}

union Frag {
    v4u   u[2];
    v16bf v;
};

#ifdef USE_TDM
// Issue one TDM 2-D tile load: TILE_C rows x 256 bf16 from global into LDS,
// with 16B of LDS padding inserted after every 512B row (pad_interval=128 dw,
// pad_amount=4 dw) to match LDS_STRIDE = 264 u16.  D# layout per CDNA5 ISA §8.
__device__ __forceinline__ void tdm_load_tile(const u16* gsrc, u16* ldst) {
    u64 ga  = (u64)(uintptr_t)gsrc;
    u32 lds = (u32)(uintptr_t)ldst;             // low 32 bits of generic ptr = LDS byte offset
    v4u g0;
    g0[0] = 1u;                                  // count=1, user descriptor
    g0[1] = lds;                                 // lds_addr [63:32]
    g0[2] = (u32)(ga & 0xFFFFFFFFu);             // global_addr [95:64]
    g0[3] = (u32)((ga >> 32) & 0x01FFFFFFu)      // global_addr [120:96]
          | (2u << 30);                          // type=2 ("image")
    v8i g1;
    g1[0] = (int)((1u << 16)                     // data_size = 1 -> 2 bytes
                | (1u << 20)                     // pad_enable
                | (6u << 22)                     // pad_interval: 128 DWORDs (512B)
                | (3u << 25));                   // pad_amount: 4 DWORDs (16B)
    g1[1] = (int)(256u << 16);                   // tensor_dim0[15:0] = 256
    g1[2] = (int)(((u32)TILE_C) << 16);          // tensor_dim0 hi = 0, tensor_dim1[15:0] = 32
    g1[3] = (int)(256u << 16);                   // tensor_dim1 hi = 0, tile_dim0 = 256
    g1[4] = (int)TILE_C;                         // tile_dim1 = 32, tile_dim2 = 0
    g1[5] = 256;                                 // tensor_dim0_stride (elements)
    g1[6] = 0;
    g1[7] = 0;
    v4i z4 = {0, 0, 0, 0};
#if defined(__clang_major__) && __clang_major__ >= 23
    v8i z8 = {0, 0, 0, 0, 0, 0, 0, 0};
    __builtin_amdgcn_tensor_load_to_lds(g0, g1, z4, z4, z8, 0);
#else
    __builtin_amdgcn_tensor_load_to_lds(g0, g1, z4, z4, 0);
#endif
}
#endif // USE_TDM

// ---------------------------------------------------------------------------
// 1) codebook: f32 -> bf16, plus per-code squared L2 norm. One wave per code.
// ---------------------------------------------------------------------------
__global__ __launch_bounds__(32) void conv_emb_kernel(const float* __restrict__ emb,
                                                      u16* __restrict__ eb,
                                                      float* __restrict__ enorm) {
    const int k = blockIdx.x;
    const int l = threadIdx.x;                  // 0..31
    const float* src = emb + k * D_;
    u16* dst = eb + k * D_;
    float s = 0.f;
#pragma unroll
    for (int i = 0; i < D_ / 32; ++i) {
        float f = src[l + i * 32];
        s += f * f;
        dst[l + i * 32] = f2bf(f);
    }
#pragma unroll
    for (int off = 16; off >= 1; off >>= 1) s += __shfl_xor(s, off, 32);
    if (l == 0) enorm[k] = s;
}

// ---------------------------------------------------------------------------
// 2) z_e (B,D,T) f32 -> x_bf16 (N = B*T, D) row-major, LDS-tiled transpose.
//    grid (T/32, D/32, B), block (32,8)
// ---------------------------------------------------------------------------
__global__ __launch_bounds__(256) void conv_x_kernel(const float* __restrict__ z,
                                                     u16* __restrict__ xb) {
    __shared__ float tile[32][33];
    const int b  = blockIdx.z;
    const int dt = blockIdx.y;
    const int tt = blockIdx.x;
    const int tx = threadIdx.x;
    const int ty = threadIdx.y;
    const float* src = z + (size_t)(b * D_ + dt * 32) * T_ + tt * 32;
#pragma unroll
    for (int i = 0; i < 4; ++i)
        tile[ty + i * 8][tx] = src[(size_t)(ty + i * 8) * T_ + tx];
    __syncthreads();
    u16* dst = xb + (size_t)(b * T_ + tt * 32) * D_ + dt * 32;
#pragma unroll
    for (int i = 0; i < 4; ++i) {
        int t = ty + i * 8;
        dst[(size_t)t * D_ + tx] = f2bf(tile[tx][t]);
    }
}

// ---------------------------------------------------------------------------
// 3) Fused GEMM + argmin.  Per wave: 16 rows resident in VGPRs; codebook
//    streamed through double-buffered LDS tiles via the Tensor Data Mover
//    (async, TENSORcnt) while WMMAs run on the other buffer.
//    score = |e|^2 - 2 * <x, e>  (|x|^2 is row-constant, dropped for argmin).
// ---------------------------------------------------------------------------
__global__ __launch_bounds__(256) void vq_argmin_kernel(const u16* __restrict__ xb,
                                                        const u16* __restrict__ eb,
                                                        const float* __restrict__ enorm,
                                                        int* __restrict__ idxp) {
    __shared__ u16 lemb[2][TILE_C * LDS_STRIDE];  // 2 x 16.5 KB, padded rows

    const int tid  = threadIdx.x;
    const int lane = tid & 31;
    const int wave = tid >> 5;
    const int hi   = lane >> 4;                   // 0: lanes 0-15, 1: lanes 16-31
    const int l15  = lane & 15;
    const int n0   = (blockIdx.x * WAVES + wave) * 16;

    // --- load the wave's 16-row A panel (16x256 bf16) into 8 fragments -----
    // A 16x32 bf16 layout: lane<16 -> M=lane, K 0..7 & 16..23 ; lane>=16 -> K 8..15 & 24..31
    Frag a[8];
    const u16* xrow = xb + (size_t)(n0 + l15) * D_;
#pragma unroll
    for (int dd = 0; dd < 8; ++dd) {
        int base = dd * 32 + hi * 8;
        a[dd].u[0] = *(const v4u*)(xrow + base);
        a[dd].u[1] = *(const v4u*)(xrow + base + 16);
    }

    float minv[8];
    int   mini[8];
#pragma unroll
    for (int r = 0; r < 8; ++r) { minv[r] = 3.4e38f; mini[r] = 0; }

#ifdef USE_TDM
    // prologue: DMA first tile into buffer 0
    if (wave == 0) {
        tdm_load_tile(eb, &lemb[0][0]);
        __builtin_amdgcn_s_wait_tensorcnt(0);
    }
    __syncthreads();
#endif

    for (int stage = 0; stage < NSTAGE; ++stage) {
        const int buf   = stage & 1;
        const int cbase = stage * TILE_C;

#ifdef USE_TDM
        // kick DMA for the next tile into the other buffer (consumers of that
        // buffer all passed the previous end-of-stage barrier)
        if (wave == 0 && stage + 1 < NSTAGE)
            tdm_load_tile(eb + (size_t)(cbase + TILE_C) * D_, &lemb[buf ^ 1][0]);
#else
        __syncthreads();
        // cooperative load: 32 codes x 512B, 16B chunks, fully coalesced
        {
            const u16* esrc = eb + (size_t)cbase * D_;
#pragma unroll
            for (int i = 0; i < 4; ++i) {
                int c    = tid + i * 256;
                int code = c >> 5;
                int co   = (c & 31) * 8;
                *(v4u*)&lemb[buf][code * LDS_STRIDE + co] = *(const v4u*)(esrc + code * D_ + co);
            }
        }
        __syncthreads();
#endif
        if (stage + 2 < NSTAGE && tid < 128)       // warm GL2 two stages ahead
            __builtin_prefetch(eb + (size_t)(cbase + 2 * TILE_C) * D_ + tid * 64, 0, 1);

        // two independent 16-code accumulator chains (32 codes per stage)
        v8f acc0 = {0.f, 0.f, 0.f, 0.f, 0.f, 0.f, 0.f, 0.f};
        v8f acc1 = {0.f, 0.f, 0.f, 0.f, 0.f, 0.f, 0.f, 0.f};
#pragma unroll
        for (int dd = 0; dd < 8; ++dd) {
            // B 32x16 bf16 layout: lane<16 -> N=lane, K 0..7 & 16..23 ; lane>=16 -> K 8..15 & 24..31
            const int off = dd * 32 + hi * 8;
            Frag b0, b1;
            const u16* p0 = &lemb[buf][l15 * LDS_STRIDE + off];
            const u16* p1 = &lemb[buf][(16 + l15) * LDS_STRIDE + off];
            b0.u[0] = *(const v4u*)(p0);
            b0.u[1] = *(const v4u*)(p0 + 16);
            b1.u[0] = *(const v4u*)(p1);
            b1.u[1] = *(const v4u*)(p1 + 16);
            acc0 = __builtin_amdgcn_wmma_f32_16x16x32_bf16(
                false, a[dd].v, false, b0.v, (short)0, acc0, false, false);
            acc1 = __builtin_amdgcn_wmma_f32_16x16x32_bf16(
                false, a[dd].v, false, b1.v, (short)0, acc1, false, false);
        }
        const float nrm0 = enorm[cbase + l15];
        const float nrm1 = enorm[cbase + 16 + l15];
        const int code0 = cbase + l15;
        const int code1 = cbase + 16 + l15;
        // D layout: VGPR r -> row (r + 8*hi), column l15
#pragma unroll
        for (int r = 0; r < 8; ++r) {
            float s0 = fmaf(-2.f, acc0[r], nrm0);
            if (s0 < minv[r]) { minv[r] = s0; mini[r] = code0; }
            float s1 = fmaf(-2.f, acc1[r], nrm1);
            if (s1 < minv[r]) { minv[r] = s1; mini[r] = code1; }
        }

#ifdef USE_TDM
        // drain the in-flight DMA, then make the new buffer visible to all waves
        if (wave == 0 && stage + 1 < NSTAGE)
            __builtin_amdgcn_s_wait_tensorcnt(0);
        __syncthreads();
#endif
    }

    // reduce argmin across the 16 columns (lanes within each half-wave)
#pragma unroll
    for (int r = 0; r < 8; ++r) {
        float v  = minv[r];
        int   ix = mini[r];
#pragma unroll
        for (int off = 8; off >= 1; off >>= 1) {
            float ov = __shfl_xor(v, off, 16);
            int   oi = __shfl_xor(ix, off, 16);
            if (ov < v || (ov == v && oi < ix)) { v = ov; ix = oi; }
        }
        if (l15 == 0) idxp[n0 + hi * 8 + r] = ix;
    }
}

// ---------------------------------------------------------------------------
// 4) Gather z_q -> out (B,D,T), accumulate commitment loss + code histogram.
//    grid (T/32, D/32, B), block (32,8)
// ---------------------------------------------------------------------------
__global__ __launch_bounds__(256) void gather_out_kernel(const float* __restrict__ z,
                                                         const float* __restrict__ emb,
                                                         const int* __restrict__ idxp,
                                                         float* __restrict__ out,
                                                         float* __restrict__ counts,
                                                         float* __restrict__ loss) {
    const int b  = blockIdx.z;
    const int dt = blockIdx.y;
    const int tt = blockIdx.x;
    const int tx = threadIdx.x;                 // t
    const int ty = threadIdx.y;                 // d sub-row
    const int t  = tt * 32 + tx;
    const int n  = b * T_ + t;
    const int code = idxp[n];
    if (blockIdx.y == 0 && ty == 0) atomicAdd(&counts[code], 1.0f);
    float ls = 0.f;
#pragma unroll
    for (int i = 0; i < 4; ++i) {
        const int d = dt * 32 + ty + i * 8;
        float e = emb[(size_t)code * D_ + d];
        size_t o = (size_t)(b * D_ + d) * T_ + t;
        float x = z[o];
        out[o] = e;                              // z_q_st == z_q numerically
        float df = e - x;
        ls += df * df;
    }
#pragma unroll
    for (int off = 16; off >= 1; off >>= 1) ls += __shfl_xor(ls, off, 32);
    __shared__ float red[8];
    if (tx == 0) red[ty] = ls;
    __syncthreads();
    if (tx == 0 && ty == 0) {
        float s = 0.f;
#pragma unroll
        for (int j = 0; j < 8; ++j) s += red[j];
        atomicAdd(loss, s);
    }
}

// ---------------------------------------------------------------------------
// 5) Finalize: vq_loss and perplexity scalars.
// ---------------------------------------------------------------------------
__global__ __launch_bounds__(256) void finalize_kernel(const float* __restrict__ counts,
                                                       const float* __restrict__ loss,
                                                       float* __restrict__ scalars) {
    __shared__ float red[256];
    const int tid = threadIdx.x;
    float s = 0.f;
    for (int k = tid; k < K_; k += 256) {
        float p = counts[k] * (1.0f / (float)N_);
        s += p * logf(p + 1e-10f);
    }
    red[tid] = s;
    __syncthreads();
    for (int st = 128; st > 0; st >>= 1) {
        if (tid < st) red[tid] += red[tid + st];
        __syncthreads();
    }
    if (tid == 0) {
        scalars[0] = 0.25f * loss[0] / (float)((size_t)N_ * D_);  // ALPHA*BETA*mean
        scalars[1] = expf(-red[0]);
    }
}

// ---------------------------------------------------------------------------
extern "C" void kernel_launch(void* const* d_in, const int* in_sizes, int n_in,
                              void* d_out, int out_size, void* d_ws, size_t ws_size,
                              hipStream_t stream) {
    (void)in_sizes; (void)n_in; (void)out_size; (void)ws_size;
    const float* z_e = (const float*)d_in[0];
    const float* emb = (const float*)d_in[1];
    float* out = (float*)d_out;

    char* ws = (char*)d_ws;
    u16*   xb     = (u16*)(ws + XB_OFF);
    u16*   eb     = (u16*)(ws + EB_OFF);
    float* enorm  = (float*)(ws + ENORM_OFF);
    int*   idxp   = (int*)(ws + IDX_OFF);
    float* counts = (float*)(ws + CNT_OFF);
    float* loss   = (float*)(ws + LOSS_OFF);

    // zero histogram + loss accumulator (counts and loss are adjacent)
    hipMemsetAsync(counts, 0, (K_ + 1) * sizeof(float), stream);

    conv_emb_kernel<<<K_, 32, 0, stream>>>(emb, eb, enorm);
    conv_x_kernel<<<dim3(T_ / 32, D_ / 32, B_), dim3(32, 8), 0, stream>>>(z_e, xb);
    vq_argmin_kernel<<<N_ / (16 * WAVES), 32 * WAVES, 0, stream>>>(xb, eb, enorm, idxp);
    gather_out_kernel<<<dim3(T_ / 32, D_ / 32, B_), dim3(32, 8), 0, stream>>>(
        z_e, emb, idxp, out, counts, loss);
    finalize_kernel<<<1, 256, 0, stream>>>(counts, loss, out + (size_t)B_ * D_ * T_);
}
